// QuantumAttention_2602750181722
// MI455X (gfx1250) — compile-verified, matrix operand order checked by reading, not run on hardware
//
#include <hip/hip_runtime.h>

// ---------------------------------------------------------------------------
// Types / helpers
// ---------------------------------------------------------------------------
typedef __bf16 bf16_t;
typedef __attribute__((ext_vector_type(16))) __bf16 v16bf;
typedef __attribute__((ext_vector_type(8)))  float  v8f;
typedef __attribute__((ext_vector_type(4))) unsigned int u32x4;
typedef __attribute__((ext_vector_type(8))) int          i32x8;
typedef __attribute__((ext_vector_type(4))) int          i32x4;

#if defined(__has_builtin)
#  if __has_builtin(__builtin_amdgcn_tensor_load_to_lds) && __has_builtin(__builtin_amdgcn_s_wait_tensorcnt)
#    define HAVE_TDM 1
#  else
#    define HAVE_TDM 0
#  endif
#else
#  define HAVE_TDM 0
#endif

struct Frag16 { uint4 lo, hi; };   // 16 bf16 per lane (A or B fragment)

static __device__ __forceinline__ v16bf frag_as(const Frag16& f) {
    union { Frag16 f; v16bf v; } u; u.f = f; return u.v;
}

static __device__ __forceinline__ unsigned short f2bf(float x) {
    union { float f; unsigned u; } v; v.f = x;
    unsigned r = v.u + 0x7FFFu + ((v.u >> 16) & 1u);   // round-nearest-even
    return (unsigned short)(r >> 16);
}
static __device__ __forceinline__ float bf2f(unsigned short b) {
    union { unsigned u; float f; } v; v.u = ((unsigned)b) << 16; return v.f;
}

static __device__ __forceinline__ v8f wmma_bf16(const Frag16& a, const Frag16& b, v8f c) {
    // D(16x16 f32) = A(16x32 bf16) * B(32x16 bf16) + C
    return __builtin_amdgcn_wmma_f32_16x16x32_bf16(
        false, frag_as(a), false, frag_as(b), (short)0, c, false, false);
}

// Problem constants
#define BATCH 2
#define SEQ   1024
#define EMB   2560
#define HEADS 32
#define HD    80      // head dim
#define HDP   96      // padded head dim (3 x 32 WMMA K-steps)

// ---------------------------------------------------------------------------
// fp32 -> bf16 conversion (vectorized x8)
// ---------------------------------------------------------------------------
__global__ void cvt_f32_bf16_kernel(const float* __restrict__ src,
                                    unsigned short* __restrict__ dst, int n8) {
    int i = blockIdx.x * blockDim.x + threadIdx.x;
    if (i >= n8) return;
    size_t o = (size_t)i * 8;
    float4 a = *(const float4*)(src + o);
    float4 b = *(const float4*)(src + o + 4);
    unsigned short q[8];
    q[0]=f2bf(a.x); q[1]=f2bf(a.y); q[2]=f2bf(a.z); q[3]=f2bf(a.w);
    q[4]=f2bf(b.x); q[5]=f2bf(b.y); q[6]=f2bf(b.z); q[7]=f2bf(b.w);
    *(uint4*)(dst + o) = *(const uint4*)q;
}

// ---------------------------------------------------------------------------
// Gate coefficients:  coeff[g][d] = (cos(ph)*[g00+g01] + sin(ph)*[g10+g11])*amp
// Padded to [3][96] (zeros for d>=80).  Also cos(quantum_phase[h]).
// ---------------------------------------------------------------------------
__global__ void coeffs_kernel(const float* __restrict__ gm,
                              const float* __restrict__ gph,
                              const float* __restrict__ gamp,
                              const float* __restrict__ qph,
                              float* __restrict__ coeffs,
                              float* __restrict__ cosph) {
    int t = threadIdx.x;
    if (t < 3 * HDP) {
        int g = t / HDP, d = t % HDP;
        float v = 0.f;
        if (d < HD) {
            float r0 = gm[g*4+0] + gm[g*4+1];
            float r1 = gm[g*4+2] + gm[g*4+3];
            float ph = gph[g*HD + d];
            v = (__cosf(ph) * r0 + __sinf(ph) * r1) * gamp[g*HD + d];
        }
        coeffs[t] = v;
    }
    if (t < HEADS) cosph[t] = __cosf(qph[t]);
}

// Zero the d=80..95 padding lanes of qbuf/kbuf rows.
__global__ void padfill_kernel(unsigned short* __restrict__ qbuf,
                               unsigned short* __restrict__ kbuf) {
    int r = blockIdx.x * blockDim.x + threadIdx.x;   // B*H*S rows
    if (r >= BATCH * HEADS * SEQ) return;
    uint4 z = {0u, 0u, 0u, 0u};
    char* qp = (char*)qbuf + (size_t)r * (HDP * 2) + HD * 2;
    char* kp = (char*)kbuf + (size_t)r * (HDP * 2) + HD * 2;
    *(uint4*)(qp)      = z; *(uint4*)(qp + 16) = z;
    *(uint4*)(kp)      = z; *(uint4*)(kp + 16) = z;
}

// ---------------------------------------------------------------------------
// Shared WMMA GEMM main loop: C(128x128) += A(128xK) * B(Kx128), bf16 in,
// fp32 accumulate.  256 threads = 8 waves, each wave owns a 16x128 strip
// (8 x v8f accumulators -> 8 WMMAs per A-fragment).  K-step 32.
// LDS is double-buffered: the next tile's global loads are issued before the
// current tile's WMMAs (global latency hides under matrix math), then dumped
// into the ping-pong buffer with a single barrier per iteration.
//   As/Bs: 2 x [128][40] bf16 (padded stride; fragments = 2 contiguous uint4)
// ---------------------------------------------------------------------------
#define GEMM_BUF (128 * 40)

static __device__ __forceinline__ void gemm_mainloop(
    const unsigned short* __restrict__ Xg,   // M x K row-major bf16
    const unsigned short* __restrict__ Wg,   // K x N row-major bf16
    int K, int N, int m0, int n0, int tid,
    unsigned short* As, unsigned short* Bs, v8f* acc) {
    const int lane = tid & 31, wid = tid >> 5;
    const int lhalf = lane >> 4, l16 = lane & 15;
    // per-thread staging coordinates
    const int ar0 = (tid) >> 2,        ajj0 = ((tid) & 3) << 3;
    const int ar1 = (tid + 256) >> 2,  ajj1 = ((tid + 256) & 3) << 3;
    const int bj0 = (tid) >> 4,        bnn0 = ((tid) & 15) << 3;
    const int bj1 = (tid + 256) >> 4,  bnn1 = ((tid + 256) & 15) << 3;

    // prologue: stage tile 0 directly
    {
        uint4 v;
        v = *(const uint4*)(Xg + (size_t)(m0 + ar0) * K + ajj0);
        *(uint4*)((char*)As + ar0 * 80 + ajj0 * 2) = v;
        v = *(const uint4*)(Xg + (size_t)(m0 + ar1) * K + ajj1);
        *(uint4*)((char*)As + ar1 * 80 + ajj1 * 2) = v;
        uint4 w0 = *(const uint4*)(Wg + (size_t)bj0 * N + n0 + bnn0);
        uint4 w1 = *(const uint4*)(Wg + (size_t)bj1 * N + n0 + bnn1);
        const unsigned short* p0 = (const unsigned short*)&w0;
        const unsigned short* p1 = (const unsigned short*)&w1;
        #pragma unroll
        for (int t = 0; t < 8; ++t) Bs[(bnn0 + t) * 40 + bj0] = p0[t];
        #pragma unroll
        for (int t = 0; t < 8; ++t) Bs[(bnn1 + t) * 40 + bj1] = p1[t];
    }
    __syncthreads();

    const int niter = K >> 5;
    for (int i = 0; i < niter; ++i) {
        unsigned short* Ac = As + (i & 1) * GEMM_BUF;
        unsigned short* Bc = Bs + (i & 1) * GEMM_BUF;
        const bool nxt = (i + 1 < niter);
        uint4 ra0, ra1, rb0, rb1;
        if (nxt) {   // issue next tile's global loads early
            int kb = (i + 1) << 5;
            ra0 = *(const uint4*)(Xg + (size_t)(m0 + ar0) * K + kb + ajj0);
            ra1 = *(const uint4*)(Xg + (size_t)(m0 + ar1) * K + kb + ajj1);
            rb0 = *(const uint4*)(Wg + (size_t)(kb + bj0) * N + n0 + bnn0);
            rb1 = *(const uint4*)(Wg + (size_t)(kb + bj1) * N + n0 + bnn1);
        }
        if (i + 2 < niter)   // warm L2 two tiles ahead -> global_prefetch_b8
            __builtin_prefetch(Xg + (size_t)(m0 + ar0) * K + ((i + 2) << 5), 0, 1);

        // compute on current buffer
        Frag16 a;
        int arow = wid * 16 + l16;
        a.lo = *(const uint4*)((const char*)Ac + arow * 80 + (lhalf ? 16 : 0));
        a.hi = *(const uint4*)((const char*)Ac + arow * 80 + (lhalf ? 48 : 32));
        Frag16 b[8];
        #pragma unroll
        for (int c2 = 0; c2 < 8; ++c2) {
            int nc = c2 * 16 + l16;
            b[c2].lo = *(const uint4*)((const char*)Bc + nc * 80 + (lhalf ? 16 : 0));
            b[c2].hi = *(const uint4*)((const char*)Bc + nc * 80 + (lhalf ? 48 : 32));
        }
        #pragma unroll
        for (int c2 = 0; c2 < 8; ++c2)
            acc[c2] = wmma_bf16(a, b[c2], acc[c2]);

        if (nxt) {   // dump staged registers into the other buffer
            unsigned short* An = As + ((i + 1) & 1) * GEMM_BUF;
            unsigned short* Bn = Bs + ((i + 1) & 1) * GEMM_BUF;
            *(uint4*)((char*)An + ar0 * 80 + ajj0 * 2) = ra0;
            *(uint4*)((char*)An + ar1 * 80 + ajj1 * 2) = ra1;
            const unsigned short* p0 = (const unsigned short*)&rb0;
            const unsigned short* p1 = (const unsigned short*)&rb1;
            #pragma unroll
            for (int t = 0; t < 8; ++t) Bn[(bnn0 + t) * 40 + bj0] = p0[t];
            #pragma unroll
            for (int t = 0; t < 8; ++t) Bn[(bnn1 + t) * 40 + bj1] = p1[t];
        }
        __syncthreads();
    }
}

// ---------------------------------------------------------------------------
// QKV GEMM: (2048 x 2560) @ (2560 x 7680) + bias, scattered into
// q/k: [B,H,S,96] bf16 (padded)   v: [B,H,S,80] bf16
// ---------------------------------------------------------------------------
__global__ __launch_bounds__(256) void qkv_gemm_kernel(
    const unsigned short* __restrict__ X,
    const unsigned short* __restrict__ W,
    const float* __restrict__ bias,
    unsigned short* __restrict__ qbuf,
    unsigned short* __restrict__ kbuf,
    unsigned short* __restrict__ vbuf) {
    __shared__ __align__(16) unsigned short As[2 * GEMM_BUF];
    __shared__ __align__(16) unsigned short Bs[2 * GEMM_BUF];
    const int tid = threadIdx.x;
    const int lane = tid & 31, wid = tid >> 5;
    const int lhalf = lane >> 4, l16 = lane & 15;
    const int m0 = blockIdx.y * 128, n0 = blockIdx.x * 128;
    v8f acc[8];
    #pragma unroll
    for (int i = 0; i < 8; ++i) acc[i] = {};
    gemm_mainloop(X, W, EMB, 3 * EMB, m0, n0, tid, As, Bs, acc);
    #pragma unroll
    for (int c2 = 0; c2 < 8; ++c2) {
        int col = n0 + c2 * 16 + l16;
        float bv = bias[col];
        #pragma unroll
        for (int e = 0; e < 8; ++e) {
            int row = m0 + wid * 16 + e + (lhalf ? 8 : 0);
            unsigned short bf = f2bf(acc[c2][e] + bv);
            int b = row >> 10, s = row & 1023;
            if (col < EMB) {
                int h = col / HD, d = col % HD;
                qbuf[((size_t)((b * HEADS + h) * SEQ + s)) * HDP + d] = bf;
            } else if (col < 2 * EMB) {
                int c = col - EMB; int h = c / HD, d = c % HD;
                kbuf[((size_t)((b * HEADS + h) * SEQ + s)) * HDP + d] = bf;
            } else {
                int c = col - 2 * EMB; int h = c / HD, d = c % HD;
                vbuf[((size_t)((b * HEADS + h) * SEQ + s)) * HD + d] = bf;
            }
        }
    }
}

// ---------------------------------------------------------------------------
// Flash-style quantum attention.  Block = (b,h, 64 q rows); 4 waves x 16 rows.
// K tiles are staged with the Tensor Data Mover (tensor_load_to_lds, tracked
// by TENSORcnt) when available; V is staged manually (needs a transpose into
// [d][key]) with all global loads issued before the scatter stores.
// ---------------------------------------------------------------------------
__global__ __launch_bounds__(128) void attn_kernel(
    const unsigned short* __restrict__ qbuf,
    const unsigned short* __restrict__ kbuf,
    const unsigned short* __restrict__ vbuf,
    const float* __restrict__ coeffs,     // [3][96]
    const float* __restrict__ cosph,      // [32]
    const float* __restrict__ istr_p,     // scalar
    unsigned short* __restrict__ attn) {  // [2048][2560] bf16
    __shared__ __align__(16) unsigned short Kt[32 * HDP];   // [key][96]
    __shared__ __align__(16) unsigned short Vt[HD * 32];    // [d][key] transposed
    __shared__ __align__(16) unsigned short Pscr[4][16 * 40];
    const int tid = threadIdx.x, lane = tid & 31, wid = tid >> 5;
    const int lhalf = lane >> 4, l16 = lane & 15;
    const int bh = blockIdx.y;
    const int h = bh & (HEADS - 1), b = bh >> 5;
    const int q0 = blockIdx.x * 64 + wid * 16;
    const float scale = 0.11180339887498949f;   // 1/sqrt(80)
    const float istr3 = istr_p[0] * (1.0f / 3.0f);

    // --- Q fragments: 4 variants x 3 K-chunks, variants scaled in-register ---
    Frag16 qf[4][3];
    const char* qrow = (const char*)(qbuf + ((size_t)bh * SEQ + q0 + l16) * HDP);
    #pragma unroll
    for (int c = 0; c < 3; ++c) {
        uint4 lo = *(const uint4*)(qrow + c * 64 + (lhalf ? 16 : 0));
        uint4 hi = *(const uint4*)(qrow + c * 64 + (lhalf ? 48 : 32));
        qf[0][c].lo = lo; qf[0][c].hi = hi;
        const unsigned short* pl = (const unsigned short*)&lo;
        const unsigned short* ph = (const unsigned short*)&hi;
        float fv[16];
        #pragma unroll
        for (int t = 0; t < 8; ++t) { fv[t] = bf2f(pl[t]); fv[8 + t] = bf2f(ph[t]); }
        int dbase = c * 32 + (lhalf ? 8 : 0);
        #pragma unroll
        for (int g = 0; g < 3; ++g) {
            unsigned short ob[16];
            #pragma unroll
            for (int t = 0; t < 8; ++t) {
                ob[t]     = f2bf(fv[t]     * coeffs[g * HDP + dbase + t]);
                ob[8 + t] = f2bf(fv[8 + t] * coeffs[g * HDP + dbase + 16 + t]);
            }
            qf[g + 1][c].lo = *(const uint4*)&ob[0];
            qf[g + 1][c].hi = *(const uint4*)&ob[8];
        }
    }

    float m_r[8], l_r[8];
    v8f o_acc[5];
    #pragma unroll
    for (int e = 0; e < 8; ++e) { m_r[e] = -3.0e38f; l_r[e] = 0.f; }
    #pragma unroll
    for (int c = 0; c < 5; ++c) o_acc[c] = {};

    const size_t kvbase = (size_t)bh * SEQ;
#if HAVE_TDM
    const unsigned ldsK = (unsigned)(size_t)(void*)&Kt[0];
#endif
    for (int kb = 0; kb < SEQ; kb += 32) {
        __syncthreads();
#if HAVE_TDM
        // --- K tile via Tensor Data Mover: 2D tile 96 x 32 rows, 2B elems ---
        if (wid == 0) {
            unsigned long long gaddr =
                (unsigned long long)(size_t)(kbuf + (kvbase + kb) * HDP);
            u32x4 g0 = { 1u,                                   // count=1
                         ldsK,                                 // lds_addr
                         (unsigned)(gaddr & 0xFFFFFFFFull),
                         (unsigned)((gaddr >> 32) & 0x1FFFFFFull) | 0x80000000u }; // type=2
            i32x8 g1 = { (int)(1u << 16),        // data_size = 2 bytes
                         (int)(96u << 16),       // tensor_dim0 = 96
                         (int)(1024u << 16),     // tensor_dim1 = 1024
                         (int)(96u << 16),       // tile_dim0 = 96
                         32,                     // tile_dim1 = 32
                         96,                     // tensor_dim0_stride = 96
                         0, 0 };
            i32x4 gz = { 0, 0, 0, 0 };
#if __clang_major__ >= 23
            i32x8 gz8 = { 0, 0, 0, 0, 0, 0, 0, 0 };
            __builtin_amdgcn_tensor_load_to_lds(g0, g1, gz, gz, gz8, 0);
#else
            __builtin_amdgcn_tensor_load_to_lds(g0, g1, gz, gz, 0);
#endif
        }
#else
        // --- K tile manual staging [32][96]: 384 uint4 by 128 threads ---
        {
            uint4 rk[3];
            #pragma unroll
            for (int it = 0; it < 3; ++it) {
                int idx = tid + it * 128;
                rk[it] = *(const uint4*)(kbuf + (kvbase + kb + idx / 12) * HDP + (idx % 12) * 8);
            }
            #pragma unroll
            for (int it = 0; it < 3; ++it) {
                int idx = tid + it * 128;
                *(uint4*)((char*)Kt + (idx / 12) * 192 + (idx % 12) * 16) = rk[it];
            }
        }
#endif
        // --- V tile transposed [d][key]: loads batched before scatter ---
        {
            uint4 rv[3];
            #pragma unroll
            for (int it = 0; it < 3; ++it) {
                int idx = tid + it * 128;
                if (idx < 320)
                    rv[it] = *(const uint4*)(vbuf + (kvbase + kb + idx / 10) * HD + (idx % 10) * 8);
            }
            #pragma unroll
            for (int it = 0; it < 3; ++it) {
                int idx = tid + it * 128;
                if (idx < 320) {
                    int key = idx / 10, jj = (idx % 10) * 8;
                    const unsigned short* pv = (const unsigned short*)&rv[it];
                    #pragma unroll
                    for (int t = 0; t < 8; ++t) Vt[(jj + t) * 32 + key] = pv[t];
                }
            }
        }
        if (kb + 32 < SEQ)   // warm L2 for next V tile
            __builtin_prefetch(vbuf + (kvbase + kb + 32 + (tid >> 2)) * HD, 0, 1);
#if HAVE_TDM
        if (wid == 0) __builtin_amdgcn_s_wait_tensorcnt(0);
#endif
        __syncthreads();

        // --- scores: 4 variants x 2 key-halves x 3 K-chunks = 24 WMMAs ---
        v8f sacc[4][2];
        #pragma unroll
        for (int g = 0; g < 4; ++g) { sacc[g][0] = {}; sacc[g][1] = {}; }
        #pragma unroll
        for (int hh = 0; hh < 2; ++hh) {
            Frag16 kf[3];
            #pragma unroll
            for (int c = 0; c < 3; ++c) {
                int key = hh * 16 + l16;
                kf[c].lo = *(const uint4*)((const char*)Kt + key * 192 + c * 64 + (lhalf ? 16 : 0));
                kf[c].hi = *(const uint4*)((const char*)Kt + key * 192 + c * 64 + (lhalf ? 48 : 32));
            }
            #pragma unroll
            for (int c = 0; c < 3; ++c)
                #pragma unroll
                for (int g = 0; g < 4; ++g)
                    sacc[g][hh] = wmma_bf16(qf[g][c], kf[c], sacc[g][hh]);
        }

        // --- interference combine + online softmax ---
        float sc[2][8], mx[8], p[2][8];
        #pragma unroll
        for (int e = 0; e < 8; ++e) mx[e] = -3.0e38f;
        #pragma unroll
        for (int hh = 0; hh < 2; ++hh)
            #pragma unroll
            for (int e = 0; e < 8; ++e) {
                float s0 = sacc[0][hh][e] * scale;
                float s1 = sacc[1][hh][e] * scale;
                float s2 = sacc[2][hh][e] * scale;
                float s3 = sacc[3][hh][e] * scale;
                float v = s0 + (s1 * s2 + s1 * s3 + s2 * s3) * istr3;
                sc[hh][e] = v;
                mx[e] = fmaxf(mx[e], v);
            }
        #pragma unroll
        for (int e = 0; e < 8; ++e)
            #pragma unroll
            for (int msk = 1; msk < 16; msk <<= 1)
                mx[e] = fmaxf(mx[e], __shfl_xor(mx[e], msk, 32));
        #pragma unroll
        for (int e = 0; e < 8; ++e) {
            float mn = fmaxf(m_r[e], mx[e]);
            float alpha = __expf(m_r[e] - mn);
            float r = 0.f;
            #pragma unroll
            for (int hh = 0; hh < 2; ++hh) {
                float pe = __expf(sc[hh][e] - mn);
                p[hh][e] = pe; r += pe;
            }
            #pragma unroll
            for (int msk = 1; msk < 16; msk <<= 1) r += __shfl_xor(r, msk, 32);
            l_r[e] = l_r[e] * alpha + r;
            m_r[e] = mn;
            #pragma unroll
            for (int c = 0; c < 5; ++c) o_acc[c][e] *= alpha;
        }

        // --- P: C-layout -> A-layout via per-wave LDS scratch, then P@V ---
        unsigned short* ps = &Pscr[wid][0];
        #pragma unroll
        for (int hh = 0; hh < 2; ++hh)
            #pragma unroll
            for (int e = 0; e < 8; ++e) {
                int row = e + (lhalf ? 8 : 0);
                ps[row * 40 + hh * 16 + l16] = f2bf(p[hh][e]);
            }
        Frag16 pa;
        pa.lo = *(const uint4*)((const char*)ps + l16 * 80 + (lhalf ? 16 : 0));
        pa.hi = *(const uint4*)((const char*)ps + l16 * 80 + (lhalf ? 48 : 32));
        Frag16 vf[5];
        #pragma unroll
        for (int c = 0; c < 5; ++c) {
            int n = c * 16 + l16;
            vf[c].lo = *(const uint4*)((const char*)Vt + n * 64 + (lhalf ? 16 : 0));
            vf[c].hi = *(const uint4*)((const char*)Vt + n * 64 + (lhalf ? 48 : 32));
        }
        #pragma unroll
        for (int c = 0; c < 5; ++c)
            o_acc[c] = wmma_bf16(pa, vf[c], o_acc[c]);
    }

    // --- finalize: O/l * cos(phase), store bf16 [B,S,E] ---
    float cph = cosph[h];
    #pragma unroll
    for (int c = 0; c < 5; ++c) {
        int col = h * HD + c * 16 + l16;
        #pragma unroll
        for (int e = 0; e < 8; ++e) {
            int s = q0 + e + (lhalf ? 8 : 0);
            float v = (o_acc[c][e] / l_r[e]) * cph;
            attn[((size_t)(b * SEQ + s)) * EMB + col] = f2bf(v);
        }
    }
}

// ---------------------------------------------------------------------------
// Output projection: (2048 x 2560) @ (2560 x 2560) + bias -> fp32 out
// ---------------------------------------------------------------------------
__global__ __launch_bounds__(256) void proj_gemm_kernel(
    const unsigned short* __restrict__ A,
    const unsigned short* __restrict__ W,
    const float* __restrict__ bias,
    float* __restrict__ out) {
    __shared__ __align__(16) unsigned short As[2 * GEMM_BUF];
    __shared__ __align__(16) unsigned short Bs[2 * GEMM_BUF];
    const int tid = threadIdx.x;
    const int lane = tid & 31, wid = tid >> 5;
    const int lhalf = lane >> 4, l16 = lane & 15;
    const int m0 = blockIdx.y * 128, n0 = blockIdx.x * 128;
    v8f acc[8];
    #pragma unroll
    for (int i = 0; i < 8; ++i) acc[i] = {};
    gemm_mainloop(A, W, EMB, EMB, m0, n0, tid, As, Bs, acc);
    #pragma unroll
    for (int c2 = 0; c2 < 8; ++c2) {
        int col = n0 + c2 * 16 + l16;
        float bv = bias[col];
        #pragma unroll
        for (int e = 0; e < 8; ++e) {
            int row = m0 + wid * 16 + e + (lhalf ? 8 : 0);
            out[(size_t)row * EMB + col] = acc[c2][e] + bv;
        }
    }
}

// ---------------------------------------------------------------------------
// Launcher
// ---------------------------------------------------------------------------
extern "C" void kernel_launch(void* const* d_in, const int* in_sizes, int n_in,
                              void* d_out, int out_size, void* d_ws, size_t ws_size,
                              hipStream_t stream) {
    const float* hidden = (const float*)d_in[0];
    const float* Wqkv   = (const float*)d_in[1];
    const float* bqkv   = (const float*)d_in[2];
    const float* Wproj  = (const float*)d_in[3];
    const float* bproj  = (const float*)d_in[4];
    const float* gm     = (const float*)d_in[5];
    const float* gph    = (const float*)d_in[6];
    const float* gamp   = (const float*)d_in[7];
    const float* qph    = (const float*)d_in[8];
    const float* istr   = (const float*)d_in[9];
    float* out = (float*)d_out;

    char* ws = (char*)d_ws;
    unsigned short* Xbf      = (unsigned short*)ws; ws += (size_t)2048 * 2560 * 2;
    unsigned short* Wqkv_bf  = (unsigned short*)ws; ws += (size_t)2560 * 7680 * 2;
    unsigned short* Wproj_bf = (unsigned short*)ws; ws += (size_t)2560 * 2560 * 2;
    unsigned short* qbuf     = (unsigned short*)ws; ws += (size_t)BATCH * HEADS * SEQ * HDP * 2;
    unsigned short* kbuf     = (unsigned short*)ws; ws += (size_t)BATCH * HEADS * SEQ * HDP * 2;
    unsigned short* vbuf     = (unsigned short*)ws; ws += (size_t)BATCH * HEADS * SEQ * HD * 2;
    unsigned short* attn     = (unsigned short*)ws; ws += (size_t)2048 * 2560 * 2;
    float* coeffs            = (float*)ws;          ws += 3 * HDP * 4;
    float* cosph             = (float*)ws;

    {   // fp32 -> bf16 conversions
        int n8 = (2048 * 2560) / 8;
        cvt_f32_bf16_kernel<<<(n8 + 255) / 256, 256, 0, stream>>>(hidden, Xbf, n8);
        n8 = (2560 * 7680) / 8;
        cvt_f32_bf16_kernel<<<(n8 + 255) / 256, 256, 0, stream>>>(Wqkv, Wqkv_bf, n8);
        n8 = (2560 * 2560) / 8;
        cvt_f32_bf16_kernel<<<(n8 + 255) / 256, 256, 0, stream>>>(Wproj, Wproj_bf, n8);
    }
    coeffs_kernel<<<1, 320, 0, stream>>>(gm, gph, gamp, qph, coeffs, cosph);
    padfill_kernel<<<(BATCH * HEADS * SEQ) / 256, 256, 0, stream>>>(qbuf, kbuf);

    qkv_gemm_kernel<<<dim3((3 * EMB) / 128, 2048 / 128), 256, 0, stream>>>(
        Xbf, Wqkv_bf, bqkv, qbuf, kbuf, vbuf);

    attn_kernel<<<dim3(SEQ / 64, BATCH * HEADS), 128, 0, stream>>>(
        qbuf, kbuf, vbuf, coeffs, cosph, istr, attn);

    proj_gemm_kernel<<<dim3(EMB / 128, 2048 / 128), 256, 0, stream>>>(
        attn, Wproj_bf, bproj, out);
}
